// NnBoard768_49555332661319
// MI455X (gfx1250) — matrix operand order repaired and checked by reading, not compiled, not run on hardware
//
#include <hip/hip_runtime.h>
#include <stdint.h>

// ---------------------------------------------------------------------------
// NNUE-style eval for MI455X (gfx1250, wave32, WMMA).
//   board (sparse, 32 ones/row) -> Linear(768,512) -> clamp01 -> Linear(1024,1) -> sigmoid
// Strategy: densify 64-position tiles into LDS, run the feature transformer as
// f16 WMMA GEMM tiles (v_wmma_f32_16x16x32_f16). Weights are staged by the
// Tensor Data Mover into a double-buffered LDS region and pipelined against
// the WMMA burst; B fragments are register double-buffered so ds_load latency
// overlaps the matrix pipe.
// ---------------------------------------------------------------------------

typedef __attribute__((ext_vector_type(16))) _Float16 v16h;
typedef __attribute__((ext_vector_type(8)))  float    v8f;
typedef __attribute__((ext_vector_type(4)))  uint32_t u32x4;
typedef __attribute__((ext_vector_type(8)))  int32_t  i32x8;
typedef __attribute__((ext_vector_type(4)))  int32_t  i32x4;

#define BATCH   32768
#define PIECES  32
#define NNZ     (BATCH * PIECES)
#define FT_OUT  512
#define NFEAT   768

#define POS_PER_WG   64            // positions per workgroup (4 M-tiles of 16)
#define KC           32            // K per WMMA chunk
#define NCHUNK       (NFEAT / KC)  // 24
#define BOARD_PITCH  772           // f32 per board row (768 + 4 pad), 16B aligned
#define STG_PITCH_H  520           // f16 per staged weight row (512 + 8 pad)
#define STG_BYTES    (KC * STG_PITCH_H * 2)              // 33280 (one buffer)
#define BOARD_OFF    (2 * STG_BYTES)                     // 66560 (double buffered)
#define BOARD_BYTES  (POS_PER_WG * BOARD_PITCH * 4)      // 197632
#define PART_OFF     (BOARD_OFF + BOARD_BYTES)           // 264192
#define SMEM_BYTES   (PART_OFF + POS_PER_WG * 4)         // 264448 (<320KB LDS/WGP)

#define WS_WT_BYTES  (NFEAT * FT_OUT * 2)                // f16 transposed weights

#if defined(__has_builtin)
#if __has_builtin(__builtin_amdgcn_tensor_load_to_lds) && \
    __has_builtin(__builtin_amdgcn_s_wait_tensorcnt)
#define USE_TDM 1
#endif
#endif
#ifndef USE_TDM
#define USE_TDM 0
#endif

// ---------------------------------------------------------------------------
// Kernel 1: transpose ft_w [512,768] f32 -> wT [768,512] f16; zero gpart[B].
// ---------------------------------------------------------------------------
__global__ void nn_prep_kernel(const float* __restrict__ ft_w,
                               _Float16* __restrict__ wT,
                               float* __restrict__ gpart) {
  int gid = blockIdx.x * blockDim.x + threadIdx.x;
  if (gid < NFEAT * FT_OUT) {
    int k = gid >> 9;          // feature index (K)
    int n = gid & 511;         // output index (N)
    wT[gid] = (_Float16)ft_w[n * NFEAT + k];
  } else {
    int p = gid - NFEAT * FT_OUT;
    if (p < BATCH) gpart[p] = 0.0f;
  }
}

#if USE_TDM
// Issue a 2D TDM tile load: wT[kc : kc+32][0 : 512] f16 -> LDS at lds_off,
// with 16B LDS padding per 1024B row (pitch 520 f16). Wave-uniform args.
__device__ __forceinline__ void tdm_issue_chunk(const _Float16* wT, int kc,
                                                uint32_t lds_off) {
  uint64_t ga = (uint64_t)(uintptr_t)wT + (uint64_t)kc * (FT_OUT * 2);
  u32x4 g0;
  g0[0] = 1u;                                              // count=1, user mode
  g0[1] = lds_off;                                         // LDS byte offset
  g0[2] = (uint32_t)ga;                                    // global_addr[31:0]
  g0[3] = (uint32_t)((ga >> 32) & 0x1FFFFFFu) | (2u << 30);// addr[56:32] | type=2
  i32x8 g1;
  g1[0] = (int)((1u << 16)      // data_size = 2 bytes
              | (1u << 20)      // pad_enable
              | (7u << 22)      // pad_interval: 256 dwords (1024B row)
              | (3u << 25));    // pad_amount: 4 dwords (16B -> pitch 520 f16)
  g1[1] = (int)(512u << 16);    // tensor_dim0 = 512
  g1[2] = (int)(32u << 16);     // tensor_dim1 = 32 (exactly the tile, no OOB)
  g1[3] = (int)(512u << 16);    // tile_dim0 = 512
  g1[4] = 32;                   // tile_dim1 = 32, tile_dim2 = 0
  g1[5] = 512;                  // tensor_dim0_stride = 512 elements
  g1[6] = 0;
  g1[7] = 0;
  i32x4 gz4; gz4[0] = 0; gz4[1] = 0; gz4[2] = 0; gz4[3] = 0;   // 2D: groups 2/3 unused
  i32x8 gz8;
  gz8[0] = 0; gz8[1] = 0; gz8[2] = 0; gz8[3] = 0;
  gz8[4] = 0; gz8[5] = 0; gz8[6] = 0; gz8[7] = 0;
  __builtin_amdgcn_tensor_load_to_lds(g0, g1, gz4, gz4, gz8, 0);
}
#else
// Fallback: cooperative staged copy with the same padded layout.
__device__ __forceinline__ void stage_copy_chunk(int tid, _Float16* dst,
                                                 const _Float16* wT, int kc) {
  for (int c = tid; c < KC * (FT_OUT / 8); c += 256) {     // 16B chunks
    int row = c >> 6, cc = c & 63;
    const uint4* src = (const uint4*)(wT + (size_t)(kc + row) * FT_OUT) + cc;
    *((uint4*)(dst + row * STG_PITCH_H) + cc) = *src;
  }
}
#endif

// ---------------------------------------------------------------------------
// Kernel 2: fused scatter + WMMA feature transformer + clamp + out_w dot.
// grid = (BATCH/POS_PER_WG, 2 sides), block = 256 threads (8 waves).
// wave w: mt = w>>1 (16-row M-tile), ns = w&1 (256-wide N slab).
// ---------------------------------------------------------------------------
__global__ void __launch_bounds__(256, 1)
nn_ft_wmma_kernel(const int* __restrict__ stm_idx,
                  const int* __restrict__ nstm_idx,
                  const float* __restrict__ values,
                  const _Float16* __restrict__ wT,
                  const float* __restrict__ ft_b,
                  const float* __restrict__ out_w,
                  float* __restrict__ gpart) {
  extern __shared__ char smem[];
  float* board = (float*)(smem + BOARD_OFF);       // densified board tile, f32
  float* part  = (float*)(smem + PART_OFF);        // per-position partial output

  const int p0   = blockIdx.x * POS_PER_WG;
  const int side = blockIdx.y;                     // 0 = stm, 1 = nstm
  const int tid  = threadIdx.x;
  const int wv   = tid >> 5;
  const int lane = tid & 31;
  const int mt   = wv >> 1;                        // 0..3: M tile
  const int ns   = wv & 1;                         // 0..1: N slab (256 wide)
  const int half = lane >> 4;                      // A/C: lanes 16-31 = +8 rows / K hi
  const int nL   = lane & 15;                      // A: row M; B/C/D: column N

  // ---- zero board + partials ----
  for (int i = tid; i < POS_PER_WG * BOARD_PITCH; i += 256) board[i] = 0.0f;
  if (tid < POS_PER_WG) part[tid] = 0.0f;
  __syncthreads();

  // ---- scatter-add sparse features (values are 1.0, collisions accumulate) ----
  const int* cols  = (side ? nstm_idx : stm_idx) + NNZ;    // row 1 of [2,NNZ]
  const int  gbase = p0 * PIECES;
  for (int i = tid; i < POS_PER_WG * PIECES; i += 256) {
    int g = gbase + i;
    int c = cols[g];
    if ((unsigned)c < (unsigned)NFEAT)
      atomicAdd(&board[(i >> 5) * BOARD_PITCH + c], values[g]);
  }

  // ---- prologue: stage K-chunk 0 into staging buffer 0 ----
#if USE_TDM
  if (wv == 0) {
    tdm_issue_chunk(wT, 0, 0u);
    __builtin_amdgcn_s_wait_tensorcnt(0);
  }
#else
  stage_copy_chunk(tid, (_Float16*)smem, wT, 0);
#endif
  __syncthreads();   // board scattered + chunk 0 staged

  v8f acc[16] = {};  // 16 N-tiles x 8 rows -> 16x256 f32 output slab per wave

  for (int ic = 0; ic < NCHUNK; ++ic) {
    const int kc = ic * KC;
    const char* cur  = smem + (uint32_t)(ic & 1) * STG_BYTES;
    const uint32_t nxt_off = (uint32_t)((ic + 1) & 1) * STG_BYTES;

    // kick off DMA for the next chunk; it overlaps this chunk's WMMA burst
#if USE_TDM
    if (wv == 0 && ic + 1 < NCHUNK) tdm_issue_chunk(wT, kc + KC, nxt_off);
#else
    if (ic + 1 < NCHUNK)
      stage_copy_chunk(tid, (_Float16*)(smem + nxt_off), wT, kc + KC);
#endif

    // ---- A fragment: 16x32 f16, lane=M (nL), half selects K subgroup ----
    const float* rp = board + (mt * 16 + nL) * BOARD_PITCH + kc + 8 * half;
    float4 x0 = *(const float4*)(rp);
    float4 x1 = *(const float4*)(rp + 4);
    float4 y0 = *(const float4*)(rp + 16);
    float4 y1 = *(const float4*)(rp + 20);
    v16h a;
    a[0]  = (_Float16)x0.x; a[1]  = (_Float16)x0.y; a[2]  = (_Float16)x0.z; a[3]  = (_Float16)x0.w;
    a[4]  = (_Float16)x1.x; a[5]  = (_Float16)x1.y; a[6]  = (_Float16)x1.z; a[7]  = (_Float16)x1.w;
    a[8]  = (_Float16)y0.x; a[9]  = (_Float16)y0.y; a[10] = (_Float16)y0.z; a[11] = (_Float16)y0.w;
    a[12] = (_Float16)y1.x; a[13] = (_Float16)y1.y; a[14] = (_Float16)y1.z; a[15] = (_Float16)y1.w;

    // ---- B fragments: 32x16 f16 tiles, lane = N column, elements = K (same order as A); 16 N-tiles per wave ----
    // Register double-buffer: loads for nt+1 issue before the WMMA for nt, so
    // the ds_load wait drops to dscnt<=16 and LDS latency hides under the XDL.
    const char* bbase = cur + (size_t)(ns * 256 + nL) * 2;
    union Frag { float4 q[2]; v16h h; };
    Frag bfr[2];
    {
      const _Float16* bp = (const _Float16*)bbase;
#pragma unroll
      for (int s = 0; s < 16; ++s)
        bfr[0].h[s] = bp[(8 * half + (s & 7) + ((s >> 3) << 4)) * STG_PITCH_H];
    }
#pragma unroll
    for (int nt = 0; nt < 16; ++nt) {
      if (nt + 1 < 16) {
        const _Float16* bp = (const _Float16*)(bbase + (nt + 1) * 32);
#pragma unroll
        for (int s = 0; s < 16; ++s)
          bfr[(nt + 1) & 1].h[s] = bp[(8 * half + (s & 7) + ((s >> 3) << 4)) * STG_PITCH_H];
      }
      acc[nt] = __builtin_amdgcn_wmma_f32_16x16x32_f16(
          false, a, false, bfr[nt & 1].h, (short)0, acc[nt], false, false);
    }

#if USE_TDM
    if (wv == 0 && ic + 1 < NCHUNK) __builtin_amdgcn_s_wait_tensorcnt(0);
#endif
    __syncthreads();   // next chunk landed, this chunk's buffer now reusable
  }

  // ---- epilogue: bias + clamp01 + dot with out_w, reduce over N ----
  float c[8];
#pragma unroll
  for (int r = 0; r < 8; ++r) c[r] = 0.0f;
#pragma unroll
  for (int nt = 0; nt < 16; ++nt) {
    int N = ns * 256 + nt * 16 + nL;
    float fb = ft_b[N];
    float ow = out_w[side * FT_OUT + N];
#pragma unroll
    for (int r = 0; r < 8; ++r) {
      float v = acc[nt][r] + fb;
      v = fminf(fmaxf(v, 0.0f), 1.0f);
      c[r] += v * ow;
    }
  }
  // reduce across the 16 lanes sharing each (row, half) — wave32 shuffles
#pragma unroll
  for (int r = 0; r < 8; ++r) {
    c[r] += __shfl_xor(c[r], 1, 32);
    c[r] += __shfl_xor(c[r], 2, 32);
    c[r] += __shfl_xor(c[r], 4, 32);
    c[r] += __shfl_xor(c[r], 8, 32);
  }
  if (nL == 0) {
#pragma unroll
    for (int r = 0; r < 8; ++r)
      atomicAdd(&part[mt * 16 + 8 * half + r], c[r]);   // 2 waves/position
  }
  __syncthreads();
  if (tid < POS_PER_WG)
    atomicAdd(&gpart[p0 + tid], part[tid]);             // combine stm + nstm WGs
}

// ---------------------------------------------------------------------------
// Kernel 3: sigmoid(out_b + partial).
// ---------------------------------------------------------------------------
__global__ void nn_finish_kernel(const float* __restrict__ gpart,
                                 const float* __restrict__ out_b,
                                 float* __restrict__ out) {
  int p = blockIdx.x * blockDim.x + threadIdx.x;
  if (p < BATCH) {
    float x = gpart[p] + out_b[0];
    out[p] = 1.0f / (1.0f + __expf(-x));
  }
}

// ---------------------------------------------------------------------------
// inputs: 0 stm_indices[2,NNZ] i32 | 1 nstm_indices[2,NNZ] i32 | 2 values[NNZ] f32
//         3 size | 4 ft_w[512,768] f32 | 5 ft_b[512] f32 | 6 out_w[1,1024] f32
//         7 out_b[1] f32           output: [B] f32
// ---------------------------------------------------------------------------
extern "C" void kernel_launch(void* const* d_in, const int* in_sizes, int n_in,
                              void* d_out, int out_size, void* d_ws, size_t ws_size,
                              hipStream_t stream) {
  (void)in_sizes; (void)n_in; (void)out_size; (void)ws_size;
  const int*   stm    = (const int*)d_in[0];
  const int*   nstm   = (const int*)d_in[1];
  const float* values = (const float*)d_in[2];
  const float* ft_w   = (const float*)d_in[4];
  const float* ft_b   = (const float*)d_in[5];
  const float* out_w  = (const float*)d_in[6];
  const float* out_b  = (const float*)d_in[7];

  _Float16* wT    = (_Float16*)d_ws;                        // 768KB f16 weights
  float*    gpart = (float*)((char*)d_ws + WS_WT_BYTES);    // 128KB partials

  int prep_threads = NFEAT * FT_OUT + BATCH;
  nn_prep_kernel<<<(prep_threads + 255) / 256, 256, 0, stream>>>(ft_w, wT, gpart);

  dim3 grid(BATCH / POS_PER_WG, 2);
  nn_ft_wmma_kernel<<<grid, 256, SMEM_BYTES, stream>>>(
      stm, nstm, values, wT, ft_b, out_w, gpart);

  nn_finish_kernel<<<(BATCH + 255) / 256, 256, 0, stream>>>(
      gpart, out_b, (float*)d_out);
}
